// WMSELoss_18193481465867
// MI455X (gfx1250) — matrix operand order repaired and checked by reading, not compiled, not run on hardware
//
#include <hip/hip_runtime.h>
#include <hip/hip_bf16.h>
#include <math.h>

typedef __attribute__((ext_vector_type(16))) _Float16 v16h;
typedef __attribute__((ext_vector_type(8)))  float    v8f;

#define ORDER   12
#define BLOCK   1024
#define NFFT    2046
#define KPAD    2048
#define HOP     511
#define WORN    1024
#define LTOT    (2048*1024)            // 2,097,152 samples
#define PADLEN  (LTOT + NFFT)          // reflect-padded length = 2,099,198
#define PADRND  2102784                // >= max tile base (2,093,056) + SPAN; mult of 256
#define NFRAMES 4104                   // LTOT//HOP
#define NTTILE  257                    // ceil(4104/16)
#define SPAN    9728                   // >= 15*511 + 2048 = 9713; mult of 8 halves
#define RROWS   22528.0f               // 2048 * 11
#define PI_F    3.14159265358979323846f

// ---------------- zero init of accumulators ----------------
__global__ __launch_bounds__(256) void zero_kernel(float* S, float* mse) {
    int i = blockIdx.x * 256 + threadIdx.x;
    if (i < 1024) { S[i] = 0.0f; mse[i] = 0.0f; }
}

// ---------------- reflect pad + f16 convert (zero tail for async staging) ----------------
__global__ __launch_bounds__(256) void pad_kernel(const float* __restrict__ x,
                                                  const float* __restrict__ y,
                                                  _Float16* __restrict__ xp,
                                                  _Float16* __restrict__ yp) {
    int i = blockIdx.x * 256 + threadIdx.x;
    if (i >= PADRND) return;
    if (i >= PADLEN) { xp[i] = (_Float16)0.0f; yp[i] = (_Float16)0.0f; return; }
    int src = i - (NFFT / 2);          // pad = 1023
    if (src < 0) src = -src;
    else if (src >= LTOT) src = 2 * LTOT - 2 - src;
    xp[i] = (_Float16)x[src];
    yp[i] = (_Float16)y[src];
}

// ---------------- DFT basis (f16, K padded to 2048) ----------------
__global__ __launch_bounds__(256) void basis_kernel(_Float16* __restrict__ cosB,
                                                    _Float16* __restrict__ sinB) {
    int gid = blockIdx.x * 256 + threadIdx.x;   // 1024*2048 entries
    int f = gid >> 11, k = gid & (KPAD - 1);
    float c = 0.0f, s = 0.0f;
    if (k < NFFT) {
        float th = (2.0f * PI_F) * (float)f * (float)k / (float)NFFT;
        __sincosf(th, &s, &c);
    }
    cosB[gid] = (_Float16)c;
    sinB[gid] = (_Float16)(-s);        // exp(-i th) = cos - i sin
}

// ---------------- Burg LPC, one workgroup per 1024-sample block ----------------
__global__ __launch_bounds__(256) void burg_kernel(const float* __restrict__ y,
                                                   float* __restrict__ a1g,
                                                   float* __restrict__ a2g) {
    int blk = blockIdx.x;
    const float* yb = y + blk * BLOCK;
    __shared__ float F[BLOCK - 1];
    __shared__ float B[BLOCK - 1];
    __shared__ float red[256];
    __shared__ float arS[ORDER + 1];
    __shared__ float denS, skS;
    int tid = threadIdx.x;

    for (int t = tid; t < BLOCK - 1; t += 256) { F[t] = yb[t + 1]; B[t] = yb[t]; }
    if (tid == 0) {
        arS[0] = 1.0f;
        for (int j = 1; j <= ORDER; j++) arS[j] = 0.0f;
    }
    __syncthreads();

    float p = 0.0f;
    for (int t = tid; t < BLOCK - 1; t += 256) p += F[t] * F[t] + B[t] * B[t];
    red[tid] = p; __syncthreads();
    for (int s = 128; s > 0; s >>= 1) { if (tid < s) red[tid] += red[tid + s]; __syncthreads(); }
    if (tid == 0) denS = red[0];
    __syncthreads();

    int foff = 0, len = BLOCK - 1;
    #pragma unroll
    for (int i = 0; i < ORDER; i++) {
        float bp = 0.0f;
        for (int t = tid; t < len; t += 256) bp += F[foff + t] * B[t];
        red[tid] = bp; __syncthreads();
        for (int s = 128; s > 0; s >>= 1) { if (tid < s) red[tid] += red[tid + s]; __syncthreads(); }
        if (tid == 0) {
            float k = -2.0f * red[0] / denS;
            skS = k;
            float tmp[ORDER + 1];
            for (int j = 1; j <= i + 1; j++) tmp[j] = arS[j] + k * arS[i + 1 - j];
            for (int j = 1; j <= i + 1; j++) arS[j] = tmp[j];
        }
        __syncthreads();
        float k = skS;
        for (int t = tid; t < len; t += 256) {
            float fv = F[foff + t], bv = B[t];
            F[foff + t] = fv + k * bv;
            B[t]        = bv + k * fv;
        }
        __syncthreads();
        if (tid == 0) {
            float bl = B[len - 1], f0 = F[foff];
            denS = (1.0f - k * k) * denS - bl * bl - f0 * f0;
        }
        foff += 1; len -= 1;
        __syncthreads();
    }
    if (tid < ORDER) {
        float g1 = powf(0.92f, (float)tid);
        float g2 = powf(0.60f, (float)tid);
        a1g[blk * ORDER + tid] = arS[tid] * g1;
        a2g[blk * ORDER + tid] = arS[tid] * g2;
    }
}

// ---------------- weight inverse sums: S[f] = sum_{b,k} |den|/|num| ----------------
__global__ __launch_bounds__(256) void weights_kernel(const float* __restrict__ a1,
                                                      const float* __restrict__ a2,
                                                      float* __restrict__ S) {
    int gid = blockIdx.x * 256 + threadIdx.x;   // 2048 * 1024
    int b = gid >> 10, f = gid & 1023;
    const float* A1 = a1 + b * ORDER;
    const float* A2 = a2 + b * ORDER;
    float omega = PI_F * (float)f / (float)WORN;
    float nr = 0.f, ni = 0.f, dr = 0.f, di = 0.f, acc = 0.f;
    #pragma unroll
    for (int j = 0; j < ORDER; j++) {
        float s, c;
        __sincosf(omega * (float)j, &s, &c);
        float w2 = A2[j], w1 = A1[j];
        nr += w2 * c;  ni -= w2 * s;    // prefix sum over j == mask k>=j
        dr += w1 * c;  di -= w1 * s;
        if (j >= 1) acc += sqrtf((dr * dr + di * di) / (nr * nr + ni * ni));
    }
    atomicAdd(&S[f], acc);
}

// ---------------- fused DFT(GEMM, WMMA) + |.| + squared-diff reduce ----------------
__global__ __launch_bounds__(256) void stft_mse_kernel(const _Float16* __restrict__ xp,
                                                       const _Float16* __restrict__ yp,
                                                       const _Float16* __restrict__ cosB,
                                                       const _Float16* __restrict__ sinB,
                                                       float* __restrict__ mseSum) {
    __shared__ __align__(16) _Float16 lx[SPAN];
    __shared__ __align__(16) _Float16 ly[SPAN];
    int t0   = blockIdx.x * 16;                 // time tile (16 frames)
    int base = t0 * HOP;                        // byte offset 2*base is 16B aligned
    int tid  = threadIdx.x;

    // Stage the 16 overlapping frames once via async global->LDS (ASYNCcnt path).
    {
        uint32_t ldsx = (uint32_t)(uintptr_t)lx;
        uint32_t ldsy = (uint32_t)(uintptr_t)ly;
        const _Float16* gx = xp + base;
        const _Float16* gy = yp + base;
        for (int c = tid; c < SPAN / 8; c += 256) {      // 16-byte chunks
            uint64_t gax = (uint64_t)(uintptr_t)(gx + c * 8);
            uint64_t gay = (uint64_t)(uintptr_t)(gy + c * 8);
            uint32_t lax = ldsx + c * 16;
            uint32_t lay = ldsy + c * 16;
            asm volatile("global_load_async_to_lds_b128 %0, %1, off"
                         :: "v"(lax), "v"(gax) : "memory");
            asm volatile("global_load_async_to_lds_b128 %0, %1, off"
                         :: "v"(lay), "v"(gay) : "memory");
        }
        asm volatile("s_wait_asynccnt 0" ::: "memory");
    }
    __syncthreads();

    int wave = tid >> 5, lane = tid & 31;
    int m = lane & 15, hi = lane >> 4;
    int tglob = t0 + m;
    float valid = (tglob < NFRAMES) ? 1.0f : 0.0f;

    for (int fi = 0; fi < 8; fi++) {
        int ftile = wave * 8 + fi;
        int frow  = ftile * 16 + m;             // A-matrix row (frequency)
        const uint32_t* crow = (const uint32_t*)(cosB + (size_t)frow * KPAD);
        const uint32_t* srow = (const uint32_t*)(sinB + (size_t)frow * KPAD);

        v8f xr = {0.f,0.f,0.f,0.f,0.f,0.f,0.f,0.f};
        v8f xi = xr, yr = xr, yi = xr;

        for (int kc = 0; kc < KPAD / 32; kc++) {
            int k0 = kc * 32;
            __builtin_prefetch((const void*)(crow + ((k0 + 32) >> 1)), 0, 3);
            __builtin_prefetch((const void*)(srow + ((k0 + 32) >> 1)), 0, 3);

            // A fragments: 16 consecutive K halves per lane half (matches B)
            union { v16h v; uint32_t u[8]; } ac, as;
            #pragma unroll
            for (int v = 0; v < 8; v++) {
                int kb = 2 * v + (hi ? 16 : 0) + k0;
                ac.u[v] = crow[kb >> 1];
                as.u[v] = srow[kb >> 1];
            }
            // B fragments: lane = time column, 16 consecutive K halves
            union { v16h v; _Float16 h[16]; } bx, by;
            int lbase = m * HOP + k0 + (hi ? 16 : 0);
            #pragma unroll
            for (int e = 0; e < 16; e++) { bx.h[e] = lx[lbase + e]; by.h[e] = ly[lbase + e]; }

            xr = __builtin_amdgcn_wmma_f32_16x16x32_f16(false, ac.v, false, bx.v, (short)0, xr, false, false);
            xi = __builtin_amdgcn_wmma_f32_16x16x32_f16(false, as.v, false, bx.v, (short)0, xi, false, false);
            yr = __builtin_amdgcn_wmma_f32_16x16x32_f16(false, ac.v, false, by.v, (short)0, yr, false, false);
            yi = __builtin_amdgcn_wmma_f32_16x16x32_f16(false, as.v, false, by.v, (short)0, yi, false, false);
        }

        // |X|,|Y| -> (|X|-|Y|)^2, reduce over the 16 time columns, accumulate per freq
        #pragma unroll
        for (int r = 0; r < 8; r++) {
            float mx = sqrtf(xr[r] * xr[r] + xi[r] * xi[r]);
            float my = sqrtf(yr[r] * yr[r] + yi[r] * yi[r]);
            float d  = (mx - my);
            d = d * d * valid;
            d += __shfl_xor(d, 1);
            d += __shfl_xor(d, 2);
            d += __shfl_xor(d, 4);
            d += __shfl_xor(d, 8);
            if ((lane & 15) == 0) {
                int fout = ftile * 16 + r + (hi ? 8 : 0);   // D layout: M = r + 8*laneHi
                atomicAdd(&mseSum[fout], d);
            }
        }
    }
}

// ---------------- final scalar: mean over (R,F) of mse[f]/W[r,f] ----------------
__global__ __launch_bounds__(256) void final_kernel(const float* __restrict__ mseSum,
                                                    const float* __restrict__ S,
                                                    float* __restrict__ out) {
    __shared__ float red[256];
    int tid = threadIdx.x;
    float p = 0.0f;
    for (int f = tid; f < 1024; f += 256)
        p += (mseSum[f] / (float)NFRAMES) * S[f];
    red[tid] = p; __syncthreads();
    for (int s = 128; s > 0; s >>= 1) { if (tid < s) red[tid] += red[tid + s]; __syncthreads(); }
    if (tid == 0) out[0] = red[0] / (RROWS * 1024.0f);
}

extern "C" void kernel_launch(void* const* d_in, const int* in_sizes, int n_in,
                              void* d_out, int out_size, void* d_ws, size_t ws_size,
                              hipStream_t stream) {
    (void)in_sizes; (void)n_in; (void)out_size; (void)ws_size;
    const float* x = (const float*)d_in[0];
    const float* y = (const float*)d_in[1];
    float* out = (float*)d_out;
    char* ws = (char*)d_ws;

    float*    a1     = (float*)(ws + 0);            // 2048*12*4      =    98304
    float*    a2     = (float*)(ws + 98304);        //                =    98304
    float*    S      = (float*)(ws + 196608);       // 1024*4
    float*    mseSum = (float*)(ws + 200704);       // 1024*4
    _Float16* cosB   = (_Float16*)(ws + 204800);    // 1024*2048*2    =  4194304
    _Float16* sinB   = (_Float16*)(ws + 4399104);   //                =  4194304
    _Float16* xp     = (_Float16*)(ws + 8593408);   // PADRND*2      =  4205568
    _Float16* yp     = (_Float16*)(ws + 12798976);  //                =  4205568
                                                    // total ~16.2 MB

    zero_kernel   <<<4,    256, 0, stream>>>(S, mseSum);
    pad_kernel    <<<PADRND / 256, 256, 0, stream>>>(x, y, xp, yp);
    basis_kernel  <<<(1024 * KPAD) / 256,  256, 0, stream>>>(cosB, sinB);
    burg_kernel   <<<2048, 256, 0, stream>>>(y, a1, a2);
    weights_kernel<<<(2048 * 1024) / 256,  256, 0, stream>>>(a1, a2, S);
    stft_mse_kernel<<<NTTILE, 256, 0, stream>>>(xp, yp, cosB, sinB, mseSum);
    final_kernel  <<<1,    256, 0, stream>>>(mseSum, S, out);
}
